// GCNClusterNet_22308060135606
// MI455X (gfx1250) — matrix operand order repaired and checked by reading, not compiled, not run on hardware
//
#include <hip/hip_runtime.h>

#define N_NODESC 50000
#define N_EDGESC 800000
#define NFEATC   512
#define NHIDC    256
#define NOUTC    128
#define KCL      50
#define KPADC    64
#define NPADC    50016   // nodes padded to multiple of 32 (1563 k-tiles)
#define TEMPC    30.0f
#define NUM_ITER 10      // harness setup fixes num_iter = 10

typedef __attribute__((ext_vector_type(16))) __bf16          v16bf;
typedef __attribute__((ext_vector_type(16))) unsigned short  v16u;
typedef __attribute__((ext_vector_type(8)))  unsigned short  v8u;
typedef __attribute__((ext_vector_type(8)))  float           v8f;

__device__ __forceinline__ unsigned short f32_to_bf16(float f) {
  unsigned int u = __float_as_uint(f);
  u += 0x7fffu + ((u >> 16) & 1u);          // round-to-nearest-even
  return (unsigned short)(u >> 16);
}

__device__ __forceinline__ float atomic_add_f32(float* p, float v) {
  return __hip_atomic_fetch_add(p, v, __ATOMIC_RELAXED, __HIP_MEMORY_SCOPE_AGENT);
}

__device__ __forceinline__ v8f wmma_bf16(v16u au, v16u bu, v8f c) {
  return __builtin_amdgcn_wmma_f32_16x16x32_bf16(
      false, __builtin_bit_cast(v16bf, au),
      false, __builtin_bit_cast(v16bf, bu),
      (short)0, c, false, false);
}

// ---------------------------------------------------------------------------
// Register-blocked WMMA inner loop: one wave computes a 16x64 f32 tile
// (4 accumulators share one A fragment per k-step -> 10 b128 loads / 4 WMMAs).
// A : row-major bf16 [M][K]   (lda = padded K)
// Bt: row-major bf16 [N][K]   (B transposed; ldb = padded K)
// Per-lane addressing (contiguous K per lane half; same order for A and B):
//   A frag: lane<16 -> row=lane, K 0..15; lane>=16 -> K 16..31
//   B frag: lane = column N%16; lane<16 -> K 0..15, lane>=16 -> K 16..31
// ---------------------------------------------------------------------------
__device__ __forceinline__ void wmma_tile_loop_x4(const unsigned short* __restrict__ A, int lda,
                                                  const unsigned short* __restrict__ Bt, int ldb,
                                                  int mt, int ntg, int kt0, int kt1,
                                                  v8f& c0, v8f& c1, v8f& c2, v8f& c3)
{
  const int lane = threadIdx.x & 31;
  const unsigned short* ap = A + (size_t)(mt * 16 + (lane & 15)) * lda + ((lane >> 4) << 4);
  const unsigned short* bp = Bt + (size_t)(ntg * 64 + (lane & 15)) * ldb + ((lane >> 4) << 4);
  const size_t bstep = (size_t)16 * ldb;
  for (int kt = kt0; kt < kt1; ++kt) {
    const unsigned short* a0 = ap + (size_t)kt * 32;
    v8u alo = *(const v8u*)(a0);            // 16B
    v8u ahi = *(const v8u*)(a0 + 8);        // 16B
    v16u au = __builtin_shufflevector(alo, ahi, 0,1,2,3,4,5,6,7,8,9,10,11,12,13,14,15);
    const unsigned short* b0 = bp + (size_t)kt * 32;
    v16u bu0 = *(const v16u*)(b0);                  // 32B contiguous
    v16u bu1 = *(const v16u*)(b0 + bstep);
    v16u bu2 = *(const v16u*)(b0 + 2 * bstep);
    v16u bu3 = *(const v16u*)(b0 + 3 * bstep);
    c0 = wmma_bf16(au, bu0, c0);
    c1 = wmma_bf16(au, bu1, c1);
    c2 = wmma_bf16(au, bu2, c2);
    c3 = wmma_bf16(au, bu3, c3);
  }
}

// C/D 16x16 f32 layout: VGPR r, lane<16 -> M=r, N=lane ; lane>=16 -> M=r+8, N=lane-16
__device__ __forceinline__ void store_tile(float* __restrict__ C, int ldc, int mt, int nt,
                                           const v8f& acc) {
  int lane = threadIdx.x & 31;
  float* cp = C + (size_t)(mt * 16 + ((lane >> 4) << 3)) * ldc + nt * 16 + (lane & 15);
#pragma unroll
  for (int r = 0; r < 8; ++r) cp[(size_t)r * ldc] = acc[r];
}

__device__ __forceinline__ void atomic_tile(float* __restrict__ C, int ldc, int mt, int nt,
                                            const v8f& acc) {
  int lane = threadIdx.x & 31;
  float* cp = C + (size_t)(mt * 16 + ((lane >> 4) << 3)) * ldc + nt * 16 + (lane & 15);
#pragma unroll
  for (int r = 0; r < 8; ++r) atomic_add_f32(cp + (size_t)r * ldc, acc[r]);
}

// nGroups = nTiles/4 ; each wave-task = (mt, group of 4 n-tiles)
__global__ void k_gemm_bf16(const unsigned short* __restrict__ A, int lda,
                            const unsigned short* __restrict__ Bt, int ldb,
                            float* __restrict__ C, int ldc,
                            int mTiles, int nGroups, int kTiles)
{
  int t = blockIdx.x * (blockDim.x >> 5) + (threadIdx.x >> 5);
  if (t >= mTiles * nGroups) return;
  int mt = t / nGroups, ntg = t - mt * nGroups;
  v8f c0 = {}, c1 = {}, c2 = {}, c3 = {};
  wmma_tile_loop_x4(A, lda, Bt, ldb, mt, ntg, 0, kTiles, c0, c1, c2, c3);
  store_tile(C, ldc, mt, ntg * 4 + 0, c0);
  store_tile(C, ldc, mt, ntg * 4 + 1, c1);
  store_tile(C, ldc, mt, ntg * 4 + 2, c2);
  store_tile(C, ldc, mt, ntg * 4 + 3, c3);
}

__global__ void k_gemm_bf16_splitk(const unsigned short* __restrict__ A, int lda,
                                   const unsigned short* __restrict__ Bt, int ldb,
                                   float* __restrict__ C, int ldc,
                                   int mTiles, int nGroups, int kTiles, int kChunk)
{
  int t = blockIdx.x * (blockDim.x >> 5) + (threadIdx.x >> 5);
  if (t >= mTiles * nGroups) return;
  int kt0 = blockIdx.y * kChunk;
  int kt1 = kt0 + kChunk; if (kt1 > kTiles) kt1 = kTiles;
  if (kt0 >= kt1) return;
  int mt = t / nGroups, ntg = t - mt * nGroups;
  v8f c0 = {}, c1 = {}, c2 = {}, c3 = {};
  wmma_tile_loop_x4(A, lda, Bt, ldb, mt, ntg, kt0, kt1, c0, c1, c2, c3);
  atomic_tile(C, ldc, mt, ntg * 4 + 0, c0);
  atomic_tile(C, ldc, mt, ntg * 4 + 1, c1);
  atomic_tile(C, ldc, mt, ntg * 4 + 2, c2);
  atomic_tile(C, ldc, mt, ntg * 4 + 3, c3);
}

// ---------------------------------------------------------------------------
// Elementwise / helper kernels
// ---------------------------------------------------------------------------
__global__ void k_f32_to_bf16(const float* __restrict__ in, unsigned short* __restrict__ out, size_t n) {
  size_t i = (size_t)blockIdx.x * blockDim.x + threadIdx.x;
  if (i < n) out[i] = f32_to_bf16(in[i]);
}

// in [R][C] f32 -> out [C][R] bf16
__global__ void k_transpose_bf16(const float* __restrict__ in, unsigned short* __restrict__ out,
                                 int R, int C) {
  int i = blockIdx.x * blockDim.x + threadIdx.x;
  if (i < R * C) {
    int r = i / C, c = i - r * C;
    out[(size_t)c * R + r] = f32_to_bf16(in[i]);
  }
}

__global__ void k_fill_u32(unsigned int* __restrict__ p, size_t n, unsigned int v) {
  size_t i = (size_t)blockIdx.x * blockDim.x + threadIdx.x;
  if (i < n) p[i] = v;
}

__global__ void k_copy_f32(float* __restrict__ dst, const float* __restrict__ src, size_t n) {
  size_t i = (size_t)blockIdx.x * blockDim.x + threadIdx.x;
  if (i < n) dst[i] = src[i];
}

// out[row[e]][f] += w[e] * H[col[e]][f]   (segment_sum scatter)
__global__ void k_spmm(const int* __restrict__ row, const int* __restrict__ col,
                       const float* __restrict__ ew, const float* __restrict__ H,
                       float* __restrict__ out, int nEdges, int fShift) {
  int F = 1 << fShift;
  int epb = blockDim.x >> fShift;
  int e = blockIdx.x * epb + (threadIdx.x >> fShift);
  if (e >= nEdges) return;
  int f = threadIdx.x & (F - 1);
  int r = row[e], c = col[e];
  float v = ew[e] * H[(size_t)c * F + f];
  atomic_add_f32(&out[(size_t)r * F + f], v);
}

__global__ void k_bias_relu_bf16(const float* __restrict__ s, const float* __restrict__ b,
                                 unsigned short* __restrict__ out, size_t n, int fMask) {
  size_t i = (size_t)blockIdx.x * blockDim.x + threadIdx.x;
  if (i < n) {
    float v = s[i] + b[i & fMask];
    out[i] = f32_to_bf16(v > 0.0f ? v : 0.0f);
  }
}

__global__ void k_bias_add(const float* __restrict__ s, const float* __restrict__ b,
                           float* __restrict__ out, size_t n, int fMask) {
  size_t i = (size_t)blockIdx.x * blockDim.x + threadIdx.x;
  if (i < n) out[i] = s[i] + b[i & fMask];
}

// row-normalize embeds; emit data_bf [node][128] and dataT_bf [128][NPADC]
__global__ void k_normalize(const float* __restrict__ emb,
                            unsigned short* __restrict__ data_bf,
                            unsigned short* __restrict__ dataT_bf) {
  int gid = blockIdx.x * blockDim.x + threadIdx.x;
  int node = gid >> 5;
  int lane = threadIdx.x & 31;
  if (node >= N_NODESC) return;
  const float* e = emb + (size_t)node * NOUTC;
  float v0 = e[lane], v1 = e[lane + 32], v2 = e[lane + 64], v3 = e[lane + 96];
  float ss = v0 * v0 + v1 * v1 + v2 * v2 + v3 * v3;
  for (int off = 16; off > 0; off >>= 1) ss += __shfl_xor(ss, off, 32);
  float inv = 1.0f / sqrtf(ss);                 // reference: no eps
  unsigned short b0 = f32_to_bf16(v0 * inv), b1 = f32_to_bf16(v1 * inv);
  unsigned short b2 = f32_to_bf16(v2 * inv), b3 = f32_to_bf16(v3 * inv);
  unsigned short* d = data_bf + (size_t)node * NOUTC;
  d[lane] = b0; d[lane + 32] = b1; d[lane + 64] = b2; d[lane + 96] = b3;
  dataT_bf[(size_t)(lane)      * NPADC + node] = b0;
  dataT_bf[(size_t)(lane + 32) * NPADC + node] = b1;
  dataT_bf[(size_t)(lane + 64) * NPADC + node] = b2;
  dataT_bf[(size_t)(lane + 96) * NPADC + node] = b3;
}

// mu [50][128] f32 -> mu_bf [64][128] bf16, rows 50..63 zero
__global__ void k_mu_pad(const float* __restrict__ mu, unsigned short* __restrict__ mu_bf) {
  int i = blockIdx.x * blockDim.x + threadIdx.x;
  if (i < KPADC * NOUTC) mu_bf[i] = (i < KCL * NOUTC) ? f32_to_bf16(mu[i]) : (unsigned short)0;
}

// per-node stable softmax over 50 clusters; write rT_bf (bf16, transposed) and
// accumulate cluster_r via LDS partials + one global atomic per block/cluster.
__global__ void k_softmax_acc(const float* __restrict__ dist,
                              unsigned short* __restrict__ rT,
                              float* __restrict__ cluster_r) {
  __shared__ float cr[KPADC];
  int tid = threadIdx.x;
  if (tid < KPADC) cr[tid] = 0.0f;
  __syncthreads();
  int node = (blockIdx.x * blockDim.x + tid) >> 5;
  int lane = tid & 31;
  if (node < N_NODESC) {
    const float* dp = dist + (size_t)node * KPADC;
    float d0 = dp[lane];
    float d1 = (lane < KCL - 32) ? dp[lane + 32] : -1e30f;
    float m = fmaxf(d0, d1);
    for (int off = 16; off > 0; off >>= 1) m = fmaxf(m, __shfl_xor(m, off, 32));
    float e0 = __expf(TEMPC * (d0 - m));
    float e1 = (lane < KCL - 32) ? __expf(TEMPC * (d1 - m)) : 0.0f;
    float s = e0 + e1;
    for (int off = 16; off > 0; off >>= 1) s += __shfl_xor(s, off, 32);
    float inv = 1.0f / s;
    float r0 = e0 * inv, r1 = e1 * inv;
    rT[(size_t)lane * NPADC + node] = f32_to_bf16(r0);
    if (lane < KCL - 32) rT[(size_t)(lane + 32) * NPADC + node] = f32_to_bf16(r1);
    atomicAdd(&cr[lane], r0);
    if (lane < KCL - 32) atomicAdd(&cr[lane + 32], r1);
  }
  __syncthreads();
  if (tid < KCL) atomic_add_f32(&cluster_r[tid], cr[tid]);
}

__global__ void k_update_mu(const float* __restrict__ mean_acc,
                            const float* __restrict__ cluster_r,
                            float* __restrict__ mu) {
  int i = blockIdx.x * blockDim.x + threadIdx.x;
  if (i < KCL * NOUTC) mu[i] = mean_acc[i] / cluster_r[i >> 7];
}

__global__ void k_final_softmax(const float* __restrict__ dist,
                                float* __restrict__ r_out,
                                float* __restrict__ dist_out) {
  int gid = blockIdx.x * blockDim.x + threadIdx.x;
  int node = gid >> 5;
  int lane = threadIdx.x & 31;
  if (node >= N_NODESC) return;
  const float* dp = dist + (size_t)node * KPADC;
  float d0 = dp[lane];
  float d1 = (lane < KCL - 32) ? dp[lane + 32] : -1e30f;
  float m = fmaxf(d0, d1);
  for (int off = 16; off > 0; off >>= 1) m = fmaxf(m, __shfl_xor(m, off, 32));
  float e0 = __expf(TEMPC * (d0 - m));
  float e1 = (lane < KCL - 32) ? __expf(TEMPC * (d1 - m)) : 0.0f;
  float s = e0 + e1;
  for (int off = 16; off > 0; off >>= 1) s += __shfl_xor(s, off, 32);
  float inv = 1.0f / s;
  float* ro = r_out + (size_t)node * KCL;
  float* dd = dist_out + (size_t)node * KCL;
  ro[lane] = e0 * inv;  dd[lane] = d0;
  if (lane < KCL - 32) { ro[lane + 32] = e1 * inv; dd[lane + 32] = d1; }
}

// ---------------------------------------------------------------------------
extern "C" void kernel_launch(void* const* d_in, const int* in_sizes, int n_in,
                              void* d_out, int out_size, void* d_ws, size_t ws_size,
                              hipStream_t stream) {
  const float* x  = (const float*)d_in[0];
  const int*  row = (const int*)d_in[1];
  const int*  col = (const int*)d_in[2];
  const float* ew = (const float*)d_in[3];
  const float* W1 = (const float*)d_in[4];
  const float* b1 = (const float*)d_in[5];
  const float* W2 = (const float*)d_in[6];
  const float* b2 = (const float*)d_in[7];
  const float* init_mu = (const float*)d_in[8];
  (void)in_sizes; (void)n_in; (void)out_size; (void)ws_size;

  float* out       = (float*)d_out;
  float* out_mu    = out;                                       // [50,128]
  float* out_r     = out_mu + (size_t)KCL * NOUTC;              // [50000,50]
  float* out_emb   = out_r + (size_t)N_NODESC * KCL;            // [50000,128]
  float* out_dist  = out_emb + (size_t)N_NODESC * NOUTC;        // [50000,50]

  unsigned char* wsp = (unsigned char*)d_ws;
  auto carve = [&](size_t bytes) -> void* {
    void* p = (void*)wsp;
    wsp += (bytes + 255) & ~(size_t)255;
    return p;
  };
  unsigned short* xb       = (unsigned short*)carve((size_t)N_NODESC * NFEATC * 2);
  unsigned short* W1t      = (unsigned short*)carve((size_t)NHIDC * NFEATC * 2);
  unsigned short* W2t      = (unsigned short*)carve((size_t)NOUTC * NHIDC * 2);
  float*          h0       = (float*)carve((size_t)N_NODESC * NHIDC * 4);
  float*          s1       = (float*)carve((size_t)N_NODESC * NHIDC * 4);
  unsigned short* hb       = (unsigned short*)carve((size_t)N_NODESC * NHIDC * 2);
  float*          h2       = (float*)carve((size_t)N_NODESC * NOUTC * 4);
  float*          s2       = (float*)carve((size_t)N_NODESC * NOUTC * 4);
  unsigned short* data_bf  = (unsigned short*)carve((size_t)N_NODESC * NOUTC * 2);
  unsigned short* dataT_bf = (unsigned short*)carve((size_t)NOUTC * NPADC * 2);
  unsigned short* rT_bf    = (unsigned short*)carve((size_t)KPADC * NPADC * 2);
  float*          dist_buf = (float*)carve((size_t)N_NODESC * KPADC * 4);
  float*          mu_buf   = (float*)carve((size_t)KCL * NOUTC * 4);
  unsigned short* mu_bf    = (unsigned short*)carve((size_t)KPADC * NOUTC * 2);
  float*          mean_acc = (float*)carve((size_t)KPADC * NOUTC * 4);
  float*          cluster_r= (float*)carve((size_t)KPADC * 4);

  const int B = 256;                    // 8 waves per block
  auto blocks = [](size_t n, int b) { return (unsigned)((n + b - 1) / b); };
  auto wave_blocks = [](size_t tasks) { return (unsigned)((tasks + 7) / 8); };

  // ---- GCN layer 1: h0 = x @ W1 (bf16 WMMA, 16x64 per wave) ---------------
  size_t nx = (size_t)N_NODESC * NFEATC;
  hipLaunchKernelGGL(k_f32_to_bf16, dim3(blocks(nx, B)), dim3(B), 0, stream, x, xb, nx);
  hipLaunchKernelGGL(k_transpose_bf16, dim3(blocks((size_t)NFEATC * NHIDC, B)), dim3(B), 0, stream,
                     W1, W1t, NFEATC, NHIDC);
  hipLaunchKernelGGL(k_transpose_bf16, dim3(blocks((size_t)NHIDC * NOUTC, B)), dim3(B), 0, stream,
                     W2, W2t, NHIDC, NOUTC);
  // tasks: 3125 mtiles x 4 ngroups = 12500 waves
  hipLaunchKernelGGL(k_gemm_bf16, dim3(wave_blocks(3125 * 4)), dim3(B), 0, stream,
                     xb, NFEATC, W1t, NFEATC, h0, NHIDC, 3125, 4, NFEATC / 32);

  // ---- SpMM 1 + bias + relu -> hb (bf16) ----------------------------------
  size_t ns1 = (size_t)N_NODESC * NHIDC;
  hipLaunchKernelGGL(k_fill_u32, dim3(blocks(ns1, B)), dim3(B), 0, stream,
                     (unsigned int*)s1, ns1, 0u);
  hipLaunchKernelGGL(k_spmm, dim3(N_EDGESC), dim3(256), 0, stream,
                     row, col, ew, h0, s1, N_EDGESC, 8);           // F=256, 1 edge/block
  hipLaunchKernelGGL(k_bias_relu_bf16, dim3(blocks(ns1, B)), dim3(B), 0, stream,
                     s1, b1, hb, ns1, NHIDC - 1);

  // ---- GCN layer 2: h2 = hb @ W2 ------------------------------------------
  hipLaunchKernelGGL(k_gemm_bf16, dim3(wave_blocks(3125 * 2)), dim3(B), 0, stream,
                     hb, NHIDC, W2t, NHIDC, h2, NOUTC, 3125, 2, NHIDC / 32);

  // ---- SpMM 2 + bias -> embeds (f32, straight to d_out) -------------------
  size_t ns2 = (size_t)N_NODESC * NOUTC;
  hipLaunchKernelGGL(k_fill_u32, dim3(blocks(ns2, B)), dim3(B), 0, stream,
                     (unsigned int*)s2, ns2, 0u);
  hipLaunchKernelGGL(k_spmm, dim3(N_EDGESC / 2), dim3(256), 0, stream,
                     row, col, ew, h2, s2, N_EDGESC, 7);           // F=128, 2 edges/block
  hipLaunchKernelGGL(k_bias_add, dim3(blocks(ns2, B)), dim3(B), 0, stream,
                     s2, b2, out_emb, ns2, NOUTC - 1);

  // ---- normalize + build data / dataT (zero pads first) -------------------
  hipLaunchKernelGGL(k_fill_u32, dim3(blocks((size_t)NOUTC * NPADC / 2, B)), dim3(B), 0, stream,
                     (unsigned int*)dataT_bf, (size_t)NOUTC * NPADC / 2, 0u);
  hipLaunchKernelGGL(k_fill_u32, dim3(blocks((size_t)KPADC * NPADC / 2, B)), dim3(B), 0, stream,
                     (unsigned int*)rT_bf, (size_t)KPADC * NPADC / 2, 0u);
  hipLaunchKernelGGL(k_normalize, dim3(6250), dim3(B), 0, stream, out_emb, data_bf, dataT_bf);

  // ---- soft k-means: 10 steps (mu_init) + 1 step (mu_final) ---------------
  hipLaunchKernelGGL(k_copy_f32, dim3(blocks((size_t)KCL * NOUTC, B)), dim3(B), 0, stream,
                     mu_buf, init_mu, (size_t)KCL * NOUTC);
  const int kTilesN = NPADC / 32;       // 1563
  const int kChunk  = 25;               // 63 k-splits
  for (int it = 0; it < NUM_ITER + 1; ++it) {
    hipLaunchKernelGGL(k_mu_pad, dim3(blocks((size_t)KPADC * NOUTC, B)), dim3(B), 0, stream,
                       mu_buf, mu_bf);
    // dist = data @ mu^T : 3125 mtiles x 1 ngroup (KPADC = 64 cols)
    hipLaunchKernelGGL(k_gemm_bf16, dim3(wave_blocks(3125)), dim3(B), 0, stream,
                       data_bf, NOUTC, mu_bf, NOUTC, dist_buf, KPADC, 3125, 1, NOUTC / 32);
    hipLaunchKernelGGL(k_fill_u32, dim3(blocks((size_t)KPADC * NOUTC, B)), dim3(B), 0, stream,
                       (unsigned int*)mean_acc, (size_t)KPADC * NOUTC, 0u);
    hipLaunchKernelGGL(k_fill_u32, dim3(1), dim3(KPADC), 0, stream,
                       (unsigned int*)cluster_r, (size_t)KPADC, 0u);
    hipLaunchKernelGGL(k_softmax_acc, dim3(6250), dim3(B), 0, stream,
                       dist_buf, rT_bf, cluster_r);
    // cluster_mean = r^T @ data : 4 mtiles x 2 ngroups, split-K over 1563 k-tiles
    hipLaunchKernelGGL(k_gemm_bf16_splitk, dim3(1, (kTilesN + kChunk - 1) / kChunk), dim3(B), 0,
                       stream, rT_bf, NPADC, dataT_bf, NPADC, mean_acc, NOUTC,
                       4, 2, kTilesN, kChunk);
    hipLaunchKernelGGL(k_update_mu, dim3(blocks((size_t)KCL * NOUTC, B)), dim3(B), 0, stream,
                       mean_acc, cluster_r, mu_buf);
  }

  // ---- final outputs: mu, dist = data@mu^T, r = softmax(T*dist) -----------
  hipLaunchKernelGGL(k_copy_f32, dim3(blocks((size_t)KCL * NOUTC, B)), dim3(B), 0, stream,
                     out_mu, mu_buf, (size_t)KCL * NOUTC);
  hipLaunchKernelGGL(k_mu_pad, dim3(blocks((size_t)KPADC * NOUTC, B)), dim3(B), 0, stream,
                     mu_buf, mu_bf);
  hipLaunchKernelGGL(k_gemm_bf16, dim3(wave_blocks(3125)), dim3(B), 0, stream,
                     data_bf, NOUTC, mu_bf, NOUTC, dist_buf, KPADC, 3125, 1, NOUTC / 32);
  hipLaunchKernelGGL(k_final_softmax, dim3(6250), dim3(B), 0, stream,
                     dist_buf, out_r, out_dist);
}
